// ReduceLayer_3650722202088
// MI455X (gfx1250) — compile-verified, matrix operand order checked by reading, not run on hardware
//
#include <hip/hip_runtime.h>

#define SEQ    2048
#define D_IN   4096
#define N_OUT  11008
#define K_TOK  2201   // int(0.2 * 11008)
#define K_CORE 4403   // int(0.4 * 11008)
#define CHUNK  (N_OUT / 256)   // 43, exact

typedef __attribute__((ext_vector_type(16))) _Float16 v16h;
typedef __attribute__((ext_vector_type(8)))  _Float16 v8h;
typedef __attribute__((ext_vector_type(8)))  float    v8f;

// ---------------------------------------------------------------------------
// fp16 -> order-preserving 16-bit key (monotone total order, matches f32 order)
// ---------------------------------------------------------------------------
__device__ __forceinline__ int f16key(unsigned int h) {
  h &= 0xFFFFu;
  return (h & 0x8000u) ? (int)(~h & 0xFFFFu) : (int)(h | 0x8000u);
}

// ---------------------------------------------------------------------------
// WMMA fragment loader: 16x32 fp16 tile, row-major, K contiguous.
// ISA layout (CDNA5 ISA): lane m(0-15) holds row M=m, K = {0..7, 16..23};
// lane m+16 holds row M=m, K = {8..15, 24..31}. Two 16B loads per lane.
// Identical layout serves A (rows of X) and B (rows of W, since B = W^T).
// ---------------------------------------------------------------------------
__device__ __forceinline__ v16h load_frag16x32(const _Float16* tile, int lane) {
  const int row  = lane & 15;
  const int koff = (lane >> 4) << 3;
  const _Float16* p = tile + (size_t)row * D_IN + koff;
  v8h lo = *(const v8h*)(p);
  v8h hi = *(const v8h*)(p + 16);
  v16h r;
#pragma unroll
  for (int i = 0; i < 8; ++i) { r[i] = lo[i]; r[i + 8] = hi[i]; }
  return r;
}

// ---------------------------------------------------------------------------
// Kernel 1: true_value = X @ W^T + bias   (fp32 accumulate, fp16 store)
// 16 waves / WG as 4x4; wave tile 64x64 (4x4 wmma frags); WG tile 256x256.
// WG-tile arithmetic intensity: 4.2 MFLOP per 32 KB unique L2 bytes per
// K-step (~131 FLOP/B); per-iter WG working set (32 KB) fits the 64 KB WGP$,
// which dedupes the 4-way A / 4-way B intra-WG refetches.
// grid = (2048/256, 11008/256) = (8, 43)
// ---------------------------------------------------------------------------
__global__ __launch_bounds__(512, 1) void sg_gemm_wmma(
    const _Float16* __restrict__ X, const _Float16* __restrict__ W,
    const _Float16* __restrict__ bias, _Float16* __restrict__ out) {
  const int lane = threadIdx.x & 31;
  const int wid  = threadIdx.x >> 5;          // 0..15
  const int wm = blockIdx.x * 256 + (wid >> 2) * 64;   // M base of wave tile
  const int wn = blockIdx.y * 256 + (wid & 3) * 64;    // N base of wave tile

  v8f acc[4][4] = {};

#pragma unroll 1
  for (int k = 0; k < D_IN; k += 32) {
    v16h a[4], b[4];
#pragma unroll
    for (int i = 0; i < 4; ++i)
      a[i] = load_frag16x32(X + (size_t)(wm + 16 * i) * D_IN + k, lane);
#pragma unroll
    for (int j = 0; j < 4; ++j)
      b[j] = load_frag16x32(W + (size_t)(wn + 16 * j) * D_IN + k, lane);
#pragma unroll
    for (int i = 0; i < 4; ++i)
#pragma unroll
      for (int j = 0; j < 4; ++j)
        acc[i][j] = __builtin_amdgcn_wmma_f32_16x16x32_f16(
            false, a[i], false, b[j], (short)0, acc[i][j], false, false);
  }

  // D layout: VGPR v, lanes 0-15 -> M=v, N=lane; lanes 16-31 -> M=v+8, N=lane-16
  const int ncol = lane & 15;
  const int mh   = (lane >> 4) << 3;
#pragma unroll
  for (int j = 0; j < 4; ++j) {
    const int col = wn + 16 * j + ncol;
    const float bv = (float)bias[col];
#pragma unroll
    for (int i = 0; i < 4; ++i) {
      const int row0 = wm + 16 * i + mh;
#pragma unroll
      for (int v = 0; v < 8; ++v)
        out[(size_t)(row0 + v) * N_OUT + col] = (_Float16)(acc[i][j][v] + bv);
    }
  }
}

// ---------------------------------------------------------------------------
// Kernel 2: per-token 2-pass radix select of the K_TOK-th largest value.
// Writes per token: thr[2*t] = threshold key T, thr[2*t+1] = r
// (r = how many key==T elements, taken in index order, are inside top-K_TOK)
// ---------------------------------------------------------------------------
__global__ __launch_bounds__(256) void sg_token_thresh(
    const unsigned short* __restrict__ tv, int* __restrict__ thr) {
  __shared__ int hist[256];
  __shared__ int s_hb, s_k2;
  const int t = threadIdx.x;
  const unsigned short* row = tv + (size_t)blockIdx.x * N_OUT;

  hist[t] = 0;
  __syncthreads();
  for (int j = t; j < N_OUT; j += 256)
    atomicAdd(&hist[f16key(row[j]) >> 8], 1);
  __syncthreads();
  if (t == 0) {
    int acc = 0, hb = 0, k2 = K_TOK;
    for (int b = 255; b >= 0; --b) {
      if (acc + hist[b] >= K_TOK) { hb = b; k2 = K_TOK - acc; break; }
      acc += hist[b];
    }
    s_hb = hb; s_k2 = k2;
  }
  __syncthreads();
  const int hb = s_hb;
  hist[t] = 0;
  __syncthreads();
  for (int j = t; j < N_OUT; j += 256) {
    const int key = f16key(row[j]);
    if ((key >> 8) == hb) atomicAdd(&hist[key & 255], 1);
  }
  __syncthreads();
  if (t == 0) {
    int acc = 0, lb = 0, r = s_k2;
    for (int b = 255; b >= 0; --b) {
      if (acc + hist[b] >= s_k2) { lb = b; r = s_k2 - acc; break; }
      acc += hist[b];
    }
    thr[blockIdx.x * 2]     = (hb << 8) | lb;
    thr[blockIdx.x * 2 + 1] = r;
  }
}

// ---------------------------------------------------------------------------
// Kernel 3: per-token counting. key>T always in top-k; key==T only the first
// r by index (exact JAX tie-break). Contiguous chunks + block prefix scan.
// ---------------------------------------------------------------------------
__global__ __launch_bounds__(256) void sg_token_count(
    const unsigned short* __restrict__ tv, const int* __restrict__ thr,
    int* __restrict__ counts) {
  __shared__ int sh[256];
  const int t = threadIdx.x;
  const int T = thr[blockIdx.x * 2];
  const int r = thr[blockIdx.x * 2 + 1];
  const unsigned short* row = tv + (size_t)blockIdx.x * N_OUT;
  const int j0 = t * CHUNK, j1 = j0 + CHUNK;

  int eq = 0;
  for (int j = j0; j < j1; ++j)
    if (f16key(row[j]) == T) ++eq;
  sh[t] = eq;
  __syncthreads();
  for (int off = 1; off < 256; off <<= 1) {
    const int v = (t >= off) ? sh[t - off] : 0;
    __syncthreads();
    sh[t] += v;
    __syncthreads();
  }
  int eqb = sh[t] - eq;  // exclusive prefix of equals (index order)
  for (int j = j0; j < j1; ++j) {
    const int key = f16key(row[j]);
    if (key > T) atomicAdd(&counts[j], 1);
    else if (key == T) { if (eqb < r) atomicAdd(&counts[j], 1); ++eqb; }
  }
}

// ---------------------------------------------------------------------------
// Kernel 4: global threshold over neuron counts (values 0..SEQ).
// ---------------------------------------------------------------------------
__global__ __launch_bounds__(256) void sg_core_thresh(
    const int* __restrict__ counts, int* __restrict__ g) {
  __shared__ int hist[SEQ + 1];
  const int t = threadIdx.x;
  for (int i = t; i <= SEQ; i += 256) hist[i] = 0;
  __syncthreads();
  for (int j = t; j < N_OUT; j += 256) {
    int c = counts[j];
    if (c < 0) c = 0; if (c > SEQ) c = SEQ;
    atomicAdd(&hist[c], 1);
  }
  __syncthreads();
  if (t == 0) {
    int acc = 0, cs = 0, r = K_CORE;
    for (int b = SEQ; b >= 0; --b) {
      if (acc + hist[b] >= K_CORE) { cs = b; r = K_CORE - acc; break; }
      acc += hist[b];
    }
    g[0] = cs; g[1] = r;
  }
}

// ---------------------------------------------------------------------------
// Kernel 5: build sorted core_idx: all count>c*, plus first r ties by index.
// Ascending-index scan -> output already sorted (matches jnp.sort(core_idx)).
// ---------------------------------------------------------------------------
__global__ __launch_bounds__(256) void sg_build_idx(
    const int* __restrict__ counts, const int* __restrict__ g,
    int* __restrict__ idx) {
  __shared__ int sgt[256], seqc[256];
  const int t = threadIdx.x;
  const int cs = g[0], r = g[1];
  const int j0 = t * CHUNK, j1 = j0 + CHUNK;

  int gt = 0, eq = 0;
  for (int j = j0; j < j1; ++j) {
    const int c = counts[j];
    gt += (c > cs); eq += (c == cs);
  }
  sgt[t] = gt; seqc[t] = eq;
  __syncthreads();
  for (int off = 1; off < 256; off <<= 1) {
    const int a = (t >= off) ? sgt[t - off] : 0;
    const int b = (t >= off) ? seqc[t - off] : 0;
    __syncthreads();
    sgt[t] += a; seqc[t] += b;
    __syncthreads();
  }
  int gtb = sgt[t] - gt;
  int eqb = seqc[t] - eq;
  for (int j = j0; j < j1; ++j) {
    const int c = counts[j];
    if (c > cs) { idx[gtb + (eqb < r ? eqb : r)] = j; ++gtb; }
    else if (c == cs) { if (eqb < r) idx[gtb + eqb] = j; ++eqb; }
  }
}

// ---------------------------------------------------------------------------
// Kernel 6: gather filtered_W rows (8 KB each, uint4 copies) + filtered_bias.
// ---------------------------------------------------------------------------
__global__ __launch_bounds__(256) void sg_gather(
    const _Float16* __restrict__ W, const _Float16* __restrict__ bias,
    const int* __restrict__ idx, _Float16* __restrict__ outW,
    _Float16* __restrict__ outB) {
  const int b = blockIdx.x;
  const int j = idx[b];
  const uint4* src = (const uint4*)(W + (size_t)j * D_IN);
  uint4* dst = (uint4*)(outW + (size_t)b * D_IN);
  for (int i = threadIdx.x; i < D_IN / 8; i += 256) dst[i] = src[i];
  if (threadIdx.x == 0) outB[b] = bias[j];
}

__global__ void sg_zero(int* __restrict__ p, int n) {
  const int i = blockIdx.x * 256 + threadIdx.x;
  if (i < n) p[i] = 0;
}

// ---------------------------------------------------------------------------
extern "C" void kernel_launch(void* const* d_in, const int* in_sizes, int n_in,
                              void* d_out, int out_size, void* d_ws, size_t ws_size,
                              hipStream_t stream) {
  (void)in_sizes; (void)n_in; (void)out_size; (void)ws_size;
  const _Float16* X  = (const _Float16*)d_in[0];
  const _Float16* Wt = (const _Float16*)d_in[1];
  const _Float16* Bv = (const _Float16*)d_in[2];

  _Float16* out_tv = (_Float16*)d_out;                       // 2048*11008
  _Float16* out_W  = out_tv + (size_t)SEQ * N_OUT;           // 4403*4096
  _Float16* out_B  = out_W + (size_t)K_CORE * D_IN;          // 4403

  char* ws = (char*)d_ws;
  int* thr    = (int*)ws;                          // 2048 * 2 ints  (16 KB)
  int* counts = (int*)(ws + 16384);                // 11008 ints     (43 KB)
  int* g      = (int*)(ws + 16384 + 44032);        // 2 ints
  int* idx    = (int*)(ws + 16384 + 44032 + 64);   // 4403 ints

  sg_zero<<<(N_OUT + 255) / 256, 256, 0, stream>>>(counts, N_OUT);

  dim3 ggrid(SEQ / 256, N_OUT / 256);              // (8, 43)
  sg_gemm_wmma<<<ggrid, 512, 0, stream>>>(X, Wt, Bv, out_tv);

  sg_token_thresh<<<SEQ, 256, 0, stream>>>((const unsigned short*)out_tv, thr);
  sg_token_count<<<SEQ, 256, 0, stream>>>((const unsigned short*)out_tv, thr, counts);
  sg_core_thresh<<<1, 256, 0, stream>>>(counts, g);
  sg_build_idx<<<1, 256, 0, stream>>>(counts, g, idx);
  sg_gather<<<K_CORE, 256, 0, stream>>>(Wt, Bv, idx, out_W, out_B);
}
